// RNNDecoder_63136019251202
// MI455X (gfx1250) — compile-verified, matrix operand order checked by reading, not run on hardware
//
#include <hip/hip_runtime.h>

// Problem constants (match the JAX reference)
#define TN 1024
#define BN 128
#define DN 128
#define HN 512
#define G4 2048          // 4*H
#define KT 640           // D + H combined GEMM depth
#define NBLK 32          // persistent workgroups (one 16-hidden-unit slice each)

typedef __attribute__((ext_vector_type(16))) __bf16 v16bf;
typedef __attribute__((ext_vector_type(8)))  __bf16 v8bf;
typedef __attribute__((ext_vector_type(8)))  float  v8f;

__device__ __forceinline__ unsigned short f32_to_bf16(float f) {
    unsigned int u = __builtin_bit_cast(unsigned int, f);
    unsigned int r = u + 0x7FFFu + ((u >> 16) & 1u);   // round-to-nearest-even
    return (unsigned short)(r >> 16);
}

__device__ __forceinline__ float fast_sigmoid(float x) {
    return __builtin_amdgcn_rcpf(1.0f + __expf(-x));
}
__device__ __forceinline__ float fast_tanh(float x) {
    x = fminf(fmaxf(x, -30.0f), 30.0f);
    float e = __expf(2.0f * x);
    return (e - 1.0f) * __builtin_amdgcn_rcpf(e + 1.0f);
}

// One 16x32 WMMA fragment chunk for this lane:
// elements 0..7 = K[base..base+8), elements 8..15 = K[base+16..base+24).
__device__ __forceinline__ v16bf load_frag(const __bf16* p) {
    v8bf lo = *(const v8bf*)(p);
    v8bf hi = *(const v8bf*)(p + 16);
    return __builtin_shufflevector(lo, hi,
            0,1,2,3,4,5,6,7,8,9,10,11,12,13,14,15);
}

// ---------------- prep kernels ----------------

__global__ void init_state_kernel(unsigned short* __restrict__ h0,
                                  unsigned int* __restrict__ syncCnt) {
    int i = blockIdx.x * blockDim.x + threadIdx.x;
    if (i < BN * HN) h0[i] = 0;
    if (i == 0) *syncCnt = 0u;
}

__global__ void convert_emb_kernel(const float* __restrict__ src,
                                   unsigned short* __restrict__ dst, int n) {
    int i = blockIdx.x * blockDim.x + threadIdx.x;
    if (i < n) dst[i] = f32_to_bf16(src[i]);
}

// Pack [w_ih | w_hh] into bf16 W[col][k], col gate-interleaved per 16 hidden units:
//   col = blk*64 + gate*16 + u  <->  n_orig = gate*512 + blk*16 + u
__global__ void pack_weights_kernel(const float* __restrict__ w_ih,
                                    const float* __restrict__ w_hh,
                                    const float* __restrict__ b_ih,
                                    const float* __restrict__ b_hh,
                                    unsigned short* __restrict__ Wp,
                                    float* __restrict__ biasp) {
    int idx = blockIdx.x * blockDim.x + threadIdx.x;
    if (idx >= G4 * KT) return;
    int col = idx / KT;
    int k   = idx - col * KT;
    int blk  = col >> 6;
    int rem  = col & 63;
    int gate = rem >> 4;
    int u    = rem & 15;
    int n_orig = gate * HN + blk * 16 + u;
    float v = (k < DN) ? w_ih[n_orig * DN + k] : w_hh[n_orig * HN + (k - DN)];
    Wp[idx] = f32_to_bf16(v);
    if (k == 0) biasp[col] = b_ih[n_orig] + b_hh[n_orig];
}

// ---------------- persistent LSTM kernel ----------------
// 32 blocks x 256 threads (8 waves), all co-resident. Block b owns hidden
// units [16b,16b+16) x all 4 gates for every batch row, across ALL timesteps:
//  - weight panel (64 packed cols x 640 K, bf16 = 80KB) staged in LDS once,
//  - cell state c and blended last_c live in registers for all 1024 steps,
//  - only h traverses global memory (double-buffered), with a device-wide
//    release/acquire barrier between steps.
__global__ __launch_bounds__(256, 1) void lstm_persistent_kernel(
    const int* __restrict__ input_ids, const int* __restrict__ seq_len,
    const __bf16* __restrict__ embw,   const __bf16* __restrict__ Wp,
    const float* __restrict__ biasp,
    unsigned short* __restrict__ hbuf0, unsigned short* __restrict__ hbuf1,
    unsigned int* __restrict__ syncCnt, float* __restrict__ out)
{
    __shared__ __align__(16) unsigned short Bsh[64 * KT];   // 80 KB

    const int tid    = threadIdx.x;
    const int lane   = tid & 31;
    const int laneM  = lane & 15;
    const int laneHi = lane >> 4;
    const int m0 = (tid >> 5) << 4;       // batch tile origin (wave id * 16)
    const int n0 = blockIdx.x << 6;       // packed gate-col origin

    // ---- stage this block's weight panel into LDS (once) ----
    {
        const v8bf* src = (const v8bf*)(Wp + (size_t)n0 * KT);
        v8bf* dst = (v8bf*)Bsh;
        const int nvec = 64 * KT / 8;                      // 5120 x 16B
#pragma unroll 4
        for (int i = tid; i < nvec; i += 256) dst[i] = src[i];
    }
    __syncthreads();

    const __bf16* pBsh[4];
    float bv[4];
#pragma unroll
    for (int nt = 0; nt < 4; ++nt) {
        pBsh[nt] = (const __bf16*)Bsh + (size_t)(nt * 16 + laneM) * KT + laneHi * 8;
        bv[nt]   = biasp[n0 + nt * 16 + laneM];
    }

    const int brow = m0 + laneM;          // batch row this lane gathers for A
    const int j    = (blockIdx.x << 4) + laneM;  // hidden unit of this lane's cols

    int   sl[8];
    float cReg[8], lcReg[8];
#pragma unroll
    for (int r = 0; r < 8; ++r) {
        sl[r]   = seq_len[m0 + (laneHi << 3) + r];
        cReg[r] = 0.0f;
        lcReg[r] = 0.0f;
    }

    for (int t = 0; t < TN; ++t) {
        const __bf16* hp = (const __bf16*)((t & 1) ? hbuf1 : hbuf0);
        unsigned short* hn = (t & 1) ? hbuf0 : hbuf1;

        const int id = input_ids[t * BN + brow];
        const __bf16* pA_emb = embw + (size_t)id   * DN + laneHi * 8;
        const __bf16* pA_h   = hp   + (size_t)brow * HN + laneHi * 8 - DN;

        v8f acc[4] = {};

        // A frag lightly pipelined (global); B frags from LDS (ds_load_b128).
        v16bf a = load_frag(pA_emb);
#pragma unroll
        for (int kk = 0; kk < KT / 32; ++kk) {
            const int k0  = kk * 32;
            v16bf an;
            if (kk < KT / 32 - 1) {
                const int k0n = k0 + 32;
                an = load_frag((k0n < DN) ? (pA_emb + k0n) : (pA_h + k0n));
            }
#pragma unroll
            for (int nt = 0; nt < 4; ++nt) {
                v16bf b = load_frag(pBsh[nt] + k0);
                acc[nt] = __builtin_amdgcn_wmma_f32_16x16x32_bf16(
                              false, a, false, b, (short)0, acc[nt], false, false);
            }
            if (kk < KT / 32 - 1) a = an;
        }

        // In-register LSTM cell: acc[0..3] = i,f,g,o of hidden unit j,
        // batch = m0 + 8*laneHi + r for accumulator register r.
#pragma unroll
        for (int r = 0; r < 8; ++r) {
            int batch = m0 + (laneHi << 3) + r;
            float iv = fast_sigmoid(acc[0][r] + bv[0]);
            float fv = fast_sigmoid(acc[1][r] + bv[1]);
            float gv = fast_tanh  (acc[2][r] + bv[2]);
            float ov = fast_sigmoid(acc[3][r] + bv[3]);
            float cN = fv * cReg[r] + iv * gv;
            float hN = ov * fast_tanh(cN);
            cReg[r] = cN;
            hn[(size_t)batch * HN + j] = f32_to_bf16(hN);
            bool hold = (t > 0) && (t >= sl[r]);
            if (!hold) lcReg[r] = cN;     // carried blend_c (masked last state)
        }

        // ---- device-wide step barrier (monotonic counter, no ABA) ----
        __builtin_amdgcn_fence(__ATOMIC_RELEASE, "agent");  // drain h stores
        __syncthreads();
        if (tid == 0) {
            __hip_atomic_fetch_add(syncCnt, 1u, __ATOMIC_RELAXED,
                                   __HIP_MEMORY_SCOPE_AGENT);
            const unsigned int target = (unsigned)NBLK * (unsigned)(t + 1);
            while (__hip_atomic_load(syncCnt, __ATOMIC_RELAXED,
                                     __HIP_MEMORY_SCOPE_AGENT) < target)
                __builtin_amdgcn_s_sleep(1);
        }
        __syncthreads();
        __builtin_amdgcn_fence(__ATOMIC_ACQUIRE, "agent"); // inv stale h lines
    }

    // Output: blended last cell state, straight from registers.
#pragma unroll
    for (int r = 0; r < 8; ++r) {
        int batch = m0 + (laneHi << 3) + r;
        out[(size_t)batch * HN + j] = lcReg[r];
    }
}

// ---------------- host launch ----------------

extern "C" void kernel_launch(void* const* d_in, const int* in_sizes, int n_in,
                              void* d_out, int out_size, void* d_ws, size_t ws_size,
                              hipStream_t stream) {
    const int*   input_ids = (const int*)d_in[0];
    const int*   seq_len   = (const int*)d_in[1];
    const float* emb_w     = (const float*)d_in[2];
    const float* w_ih      = (const float*)d_in[3];
    const float* w_hh      = (const float*)d_in[4];
    const float* b_ih      = (const float*)d_in[5];
    const float* b_hh      = (const float*)d_in[6];
    float* out = (float*)d_out;

    // Workspace layout (all offsets multiples of 256 bytes)
    char* ws = (char*)d_ws;
    size_t o = 0;
    unsigned short* emb_bf = (unsigned short*)(ws + o); o += (size_t)50257 * DN * 2;
    unsigned short* Wp     = (unsigned short*)(ws + o); o += (size_t)G4 * KT * 2;
    float*          biasp  = (float*)(ws + o);          o += (size_t)G4 * 4;
    unsigned short* h0     = (unsigned short*)(ws + o); o += (size_t)BN * HN * 2;
    unsigned short* h1     = (unsigned short*)(ws + o); o += (size_t)BN * HN * 2;
    unsigned int*   syncC  = (unsigned int*)(ws + o);   o += 256;
    (void)ws_size; (void)in_sizes; (void)n_in; (void)out_size;

    init_state_kernel<<<(BN * HN + 255) / 256, 256, 0, stream>>>(h0, syncC);
    convert_emb_kernel<<<(50257 * DN + 255) / 256, 256, 0, stream>>>(
        emb_w, emb_bf, 50257 * DN);
    pack_weights_kernel<<<(G4 * KT + 255) / 256, 256, 0, stream>>>(
        w_ih, w_hh, b_ih, b_hh, Wp, biasp);

    lstm_persistent_kernel<<<NBLK, 256, 0, stream>>>(
        input_ids, seq_len, (const __bf16*)emb_bf, (const __bf16*)Wp,
        biasp, h0, h1, syncC, out);
}
